// MoE_34308198761253
// MI455X (gfx1250) — matrix operand order repaired and checked by reading, not compiled, not run on hardware
//
#include <hip/hip_runtime.h>

// ---------------- problem constants ----------------
#define N_TOKENS 2048
#define HIDDEN   2048
#define N_EXP    8
#define TOPK     2
#define D_FF     1408

typedef __attribute__((ext_vector_type(16))) _Float16 v16h;
typedef __attribute__((ext_vector_type(4)))  _Float16 v4h;
typedef __attribute__((ext_vector_type(8)))  float    v8f;
typedef int v4i_vs __attribute__((vector_size(16)));

union F16x16 { v16h v; uint4 q[2]; };

// ---------------- workspace layout (bytes) ----------------
#define OFF_CNT  0                         // 8 * int
#define OFF_TOK  256                       // 8*2048 int   (packed token*2+k)
#define OFF_WT   (OFF_TOK + N_EXP*N_TOKENS*4)       // 8*2048 float (gate weight)
#define OFF_H16  (OFF_WT  + N_EXP*N_TOKENS*4)       // 2048*2048 f16
#define OFF_A    (OFF_H16 + (size_t)N_TOKENS*HIDDEN*2)       // [8*2048][1408] f16
#define OFF_Y    (OFF_A   + (size_t)N_EXP*N_TOKENS*D_FF*2)   // [2048*2][2048] f32

// tile config
#define BM 128
#define BN 128
#define BK 32
#define LDH 40   // LDS row stride in halves (80B, 16B-aligned, conflict-padded)

// ---------------- async global->LDS copy (CDNA5 path w/ fallback) ----------
#if defined(__has_builtin)
# if __has_builtin(__builtin_amdgcn_global_load_async_to_lds_b128) && \
     __has_builtin(__builtin_amdgcn_s_wait_asynccnt)
#  define HAVE_ASYNC_LDS 1
# endif
#endif
#ifndef HAVE_ASYNC_LDS
# define HAVE_ASYNC_LDS 0
#endif

__device__ __forceinline__ void cp16_g2s(void* lds_dst, const void* gsrc) {
#if HAVE_ASYNC_LDS
    __builtin_amdgcn_global_load_async_to_lds_b128(
        (__attribute__((address_space(1))) v4i_vs*)(uintptr_t)gsrc,
        (__attribute__((address_space(3))) v4i_vs*)(unsigned)(uintptr_t)lds_dst,
        0, 0);
#else
    *(uint4*)lds_dst = *(const uint4*)gsrc;
#endif
}
__device__ __forceinline__ void cp_g2s_wait() {
#if HAVE_ASYNC_LDS
    __builtin_amdgcn_s_wait_asynccnt(0);
#endif
}

// ---------------- kernel 0: zero expert counters ----------------
__global__ void moe_init(int* ecnt) {
    if (threadIdx.x < N_EXP) ecnt[threadIdx.x] = 0;
}

// ---------------- kernel 1: h (f32) -> h16 (f16) ----------------
__global__ void moe_h16(const float* __restrict__ h, _Float16* __restrict__ h16) {
    int g = blockIdx.x * blockDim.x + threadIdx.x;      // group of 4
    size_t i = (size_t)g * 4;
    float4 f = *(const float4*)(h + i);
    v4h o; o[0] = (_Float16)f.x; o[1] = (_Float16)f.y;
           o[2] = (_Float16)f.z; o[3] = (_Float16)f.w;
    *(v4h*)(h16 + i) = o;
}

// ---------------- kernel 2: gating (one wave32 per token) ----------------
__global__ void moe_gate(const float* __restrict__ h, const float* __restrict__ gw,
                         int* __restrict__ ecnt, int* __restrict__ etok,
                         float* __restrict__ ewt) {
    int tid  = threadIdx.x;
    int lane = tid & 31;
    int t    = blockIdx.x * (blockDim.x / 32) + (tid >> 5);
    if (t >= N_TOKENS) return;

    float acc[N_EXP];
#pragma unroll
    for (int e = 0; e < N_EXP; ++e) acc[e] = 0.f;

    for (int d = lane; d < HIDDEN; d += 32) {
        float hv = h[(size_t)t * HIDDEN + d];
#pragma unroll
        for (int e = 0; e < N_EXP; ++e) acc[e] += hv * gw[e * HIDDEN + d];
    }
#pragma unroll
    for (int off = 16; off > 0; off >>= 1)
#pragma unroll
        for (int e = 0; e < N_EXP; ++e) acc[e] += __shfl_xor(acc[e], off, 32);

    if (lane == 0) {
        float mx = acc[0];
#pragma unroll
        for (int e = 1; e < N_EXP; ++e) mx = fmaxf(mx, acc[e]);
        float s[N_EXP]; float sum = 0.f;
#pragma unroll
        for (int e = 0; e < N_EXP; ++e) { s[e] = __expf(acc[e] - mx); sum += s[e]; }
        float inv = 1.f / sum;
        int i1 = 0;
#pragma unroll
        for (int e = 1; e < N_EXP; ++e) if (s[e] > s[i1]) i1 = e;
        int i2 = (i1 == 0) ? 1 : 0;
#pragma unroll
        for (int e = 0; e < N_EXP; ++e) if (e != i1 && s[e] > s[i2]) i2 = e;

        int sel[2] = { i1, i2 };
        float w2[2] = { s[i1] * inv, s[i2] * inv };
#pragma unroll
        for (int k = 0; k < TOPK; ++k) {
            int e = sel[k];
            int slot = atomicAdd(&ecnt[e], 1);
            etok[e * N_TOKENS + slot] = t * TOPK + k;
            ewt [e * N_TOKENS + slot] = w2[k];
        }
    }
}

// ---------------- kernel 3: a = silu(h*Wg) * (h*Wu)  (grouped by expert) ----
__global__ void __launch_bounds__(256)
moe_ffn1(const _Float16* __restrict__ h16,
         const float* __restrict__ wg_all, const float* __restrict__ wu_all,
         const int* __restrict__ ecnt, const int* __restrict__ etok,
         _Float16* __restrict__ a_buf) {
    const int e   = blockIdx.z;
    const int cnt = ecnt[e];
    const int m0  = blockIdx.x * BM;
    if (m0 >= cnt) return;
    const int n0  = blockIdx.y * BN;

    __shared__ int      sh_tok[BM];
    __shared__ __align__(16) _Float16 sh_h [BM * LDH];   // [m][k]
    __shared__ __align__(16) _Float16 sh_wg[BN * LDH];   // [n][k] (transposed)
    __shared__ __align__(16) _Float16 sh_wu[BN * LDH];

    const int tid  = threadIdx.x;
    const int lane = tid & 31;
    const int w    = tid >> 5;          // wave id 0..7 -> 16-row strip
    const int l16  = lane & 15;
    const int lhi  = lane >> 4;

    if (tid < BM) {
        int slot = m0 + tid;
        sh_tok[tid] = (slot < cnt) ? etok[e * N_TOKENS + slot] : 0;
    }

    const float* wg = wg_all + (size_t)e * HIDDEN * D_FF;
    const float* wu = wu_all + (size_t)e * HIDDEN * D_FF;

    // per-thread staging roles
    const int am   = tid >> 1;          // A tile: row 0..127
    const int ah   = tid & 1;           //         k-half (16 halves each)
    const int wn   = tid & 127;         // W tile: column n 0..127
    const int wkh  = tid >> 7;          //         k-half: k = wkh*16 + 0..15

    v8f gacc[BN / 16], uacc[BN / 16];
#pragma unroll
    for (int j = 0; j < BN / 16; ++j) {
        gacc[j] = (v8f){0,0,0,0,0,0,0,0};
        uacc[j] = (v8f){0,0,0,0,0,0,0,0};
    }

    for (int k0 = 0; k0 < HIDDEN; k0 += BK) {
        __syncthreads();
        // --- stage h tile [BM x BK] (f16, k-contig) via async global->LDS ---
        {
            int t = sh_tok[am] >> 1;
            const _Float16* src = h16 + (size_t)t * HIDDEN + k0 + ah * 16;
            _Float16* dst = &sh_h[am * LDH + ah * 16];
            cp16_g2s(dst,     src);
            cp16_g2s(dst + 8, src + 8);
        }
        // --- stage Wg/Wu tiles [BK x BN] transposed to [n][k] ---
        {
            const float* gsrc = wg + (size_t)(k0 + wkh * 16) * D_FF + n0 + wn;
            const float* usrc = wu + (size_t)(k0 + wkh * 16) * D_FF + n0 + wn;
            F16x16 pg, pu;
#pragma unroll
            for (int j = 0; j < 16; ++j) {
                pg.v[j] = (_Float16)gsrc[(size_t)j * D_FF];
                pu.v[j] = (_Float16)usrc[(size_t)j * D_FF];
            }
            uint4* dg = (uint4*)&sh_wg[wn * LDH + wkh * 16];
            dg[0] = pg.q[0]; dg[1] = pg.q[1];
            uint4* du = (uint4*)&sh_wu[wn * LDH + wkh * 16];
            du[0] = pu.q[0]; du[1] = pu.q[1];
        }
        cp_g2s_wait();
        __syncthreads();

        // --- A fragment: row = w*16 + l16, K = lhi*16 .. lhi*16+15 ---
        F16x16 A;
        {
            const _Float16* base = &sh_h[(w * 16 + l16) * LDH + lhi * 16];
            A.q[0] = *(const uint4*)(base);
            A.q[1] = *(const uint4*)(base + 8);
        }
        // --- per n-subtile B fragments + WMMA ---
#pragma unroll
        for (int j = 0; j < BN / 16; ++j) {
            F16x16 Bg, Bu;
            const _Float16* bg = &sh_wg[(j * 16 + l16) * LDH + lhi * 16];
            const _Float16* bu = &sh_wu[(j * 16 + l16) * LDH + lhi * 16];
            Bg.q[0] = *(const uint4*)(bg); Bg.q[1] = *(const uint4*)(bg + 8);
            Bu.q[0] = *(const uint4*)(bu); Bu.q[1] = *(const uint4*)(bu + 8);
            gacc[j] = __builtin_amdgcn_wmma_f32_16x16x32_f16(
                false, A.v, false, Bg.v, (short)0, gacc[j], false, false);
            uacc[j] = __builtin_amdgcn_wmma_f32_16x16x32_f16(
                false, A.v, false, Bu.v, (short)0, uacc[j], false, false);
        }
    }

    // --- epilogue: silu(g)*u -> f16 a_buf ---
#pragma unroll
    for (int j = 0; j < BN / 16; ++j) {
#pragma unroll
        for (int i = 0; i < 8; ++i) {
            int row  = w * 16 + lhi * 8 + i;
            int slot = m0 + row;
            if (slot < cnt) {
                float g = gacc[j][i], u = uacc[j][i];
                float a = (g / (1.0f + __expf(-g))) * u;
                int col = n0 + j * 16 + l16;
                a_buf[((size_t)(e * N_TOKENS + slot)) * D_FF + col] = (_Float16)a;
            }
        }
    }
}

// ---------------- kernel 4: y = wgt * (a * Wd)  -> y_partial[(t*2+k)][d] ----
__global__ void __launch_bounds__(256)
moe_ffn2(const _Float16* __restrict__ a_buf, const float* __restrict__ wd_all,
         const int* __restrict__ ecnt, const int* __restrict__ etok,
         const float* __restrict__ ewt, float* __restrict__ y_partial) {
    const int e   = blockIdx.z;
    const int cnt = ecnt[e];
    const int m0  = blockIdx.x * BM;
    if (m0 >= cnt) return;
    const int n0  = blockIdx.y * BN;

    __shared__ int      sh_tok[BM];
    __shared__ float    sh_wgt[BM];
    __shared__ __align__(16) _Float16 sh_a[BM * LDH];
    __shared__ __align__(16) _Float16 sh_w[BN * LDH];

    const int tid  = threadIdx.x;
    const int lane = tid & 31;
    const int w    = tid >> 5;
    const int l16  = lane & 15;
    const int lhi  = lane >> 4;

    if (tid < BM) {
        int slot = m0 + tid;
        sh_tok[tid] = (slot < cnt) ? etok[e * N_TOKENS + slot] : 0;
        sh_wgt[tid] = (slot < cnt) ? ewt [e * N_TOKENS + slot] : 0.f;
    }

    const float* wd = wd_all + (size_t)e * D_FF * HIDDEN;

    const int am  = tid >> 1;
    const int ah  = tid & 1;
    const int wn  = tid & 127;
    const int wkh = tid >> 7;

    v8f acc[BN / 16];
#pragma unroll
    for (int j = 0; j < BN / 16; ++j) acc[j] = (v8f){0,0,0,0,0,0,0,0};

    for (int k0 = 0; k0 < D_FF; k0 += BK) {          // 44 iterations
        __syncthreads();
        {
            const _Float16* src = a_buf +
                ((size_t)(e * N_TOKENS + m0 + am)) * D_FF + k0 + ah * 16;
            _Float16* dst = &sh_a[am * LDH + ah * 16];
            cp16_g2s(dst,     src);
            cp16_g2s(dst + 8, src + 8);
        }
        {
            const float* gsrc = wd + (size_t)(k0 + wkh * 16) * HIDDEN + n0 + wn;
            F16x16 pw;
#pragma unroll
            for (int j = 0; j < 16; ++j)
                pw.v[j] = (_Float16)gsrc[(size_t)j * HIDDEN];
            uint4* dw = (uint4*)&sh_w[wn * LDH + wkh * 16];
            dw[0] = pw.q[0]; dw[1] = pw.q[1];
        }
        cp_g2s_wait();
        __syncthreads();

        F16x16 A;
        {
            const _Float16* base = &sh_a[(w * 16 + l16) * LDH + lhi * 16];
            A.q[0] = *(const uint4*)(base);
            A.q[1] = *(const uint4*)(base + 8);
        }
#pragma unroll
        for (int j = 0; j < BN / 16; ++j) {
            F16x16 B;
            const _Float16* bp = &sh_w[(j * 16 + l16) * LDH + lhi * 16];
            B.q[0] = *(const uint4*)(bp); B.q[1] = *(const uint4*)(bp + 8);
            acc[j] = __builtin_amdgcn_wmma_f32_16x16x32_f16(
                false, A.v, false, B.v, (short)0, acc[j], false, false);
        }
    }

#pragma unroll
    for (int j = 0; j < BN / 16; ++j) {
#pragma unroll
        for (int i = 0; i < 8; ++i) {
            int row  = w * 16 + lhi * 8 + i;
            int slot = m0 + row;
            if (slot < cnt) {
                int   p   = sh_tok[row];       // t*2+k
                float wgt = sh_wgt[row];
                int   col = n0 + j * 16 + l16;
                y_partial[(size_t)p * HIDDEN + col] = acc[j][i] * wgt;
            }
        }
    }
}

// ---------------- kernel 5: combine out[t] = y[t,0] + y[t,1] ----------------
__global__ void moe_combine(const float* __restrict__ yp, float* __restrict__ out) {
    int g = blockIdx.x * blockDim.x + threadIdx.x;   // group of 4 floats
    size_t i = (size_t)g * 4;
    int t = (int)(i >> 11);
    int d = (int)(i & (HIDDEN - 1));
    float4 a = *(const float4*)(yp + (size_t)t * (TOPK * HIDDEN) + d);
    float4 b = *(const float4*)(yp + (size_t)t * (TOPK * HIDDEN) + HIDDEN + d);
    float4 o; o.x = a.x + b.x; o.y = a.y + b.y; o.z = a.z + b.z; o.w = a.w + b.w;
    *(float4*)(out + i) = o;
}

// ---------------- launch ----------------
extern "C" void kernel_launch(void* const* d_in, const int* in_sizes, int n_in,
                              void* d_out, int out_size, void* d_ws, size_t ws_size,
                              hipStream_t stream) {
    const float* h  = (const float*)d_in[0];
    const float* gw = (const float*)d_in[1];
    const float* wg = (const float*)d_in[2];
    const float* wu = (const float*)d_in[3];
    const float* wd = (const float*)d_in[4];
    float* out = (float*)d_out;

    char* ws = (char*)d_ws;
    int*      ecnt = (int*)     (ws + OFF_CNT);
    int*      etok = (int*)     (ws + OFF_TOK);
    float*    ewt  = (float*)   (ws + OFF_WT);
    _Float16* h16  = (_Float16*)(ws + OFF_H16);
    _Float16* abuf = (_Float16*)(ws + OFF_A);
    float*    yp   = (float*)   (ws + OFF_Y);

    moe_init<<<1, 32, 0, stream>>>(ecnt);
    moe_h16 <<<(N_TOKENS * HIDDEN / 4) / 256, 256, 0, stream>>>(h, h16);
    moe_gate<<<N_TOKENS / 8, 256, 0, stream>>>(h, gw, ecnt, etok, ewt);

    dim3 g1(N_TOKENS / BM, D_FF / BN, N_EXP);     // 16 x 11 x 8
    moe_ffn1<<<g1, 256, 0, stream>>>(h16, wg, wu, ecnt, etok, abuf);

    dim3 g2(N_TOKENS / BM, HIDDEN / BN, N_EXP);   // 16 x 16 x 8
    moe_ffn2<<<g2, 256, 0, stream>>>(abuf, wd, ecnt, etok, ewt, yp);

    moe_combine<<<(N_TOKENS * HIDDEN / 4) / 256, 256, 0, stream>>>(yp, out);
}
